// SoftAttention_41437844472135
// MI455X (gfx1250) — matrix operand order repaired and checked by reading, not compiled, not run on hardware
//
#include <hip/hip_runtime.h>
#include <stdint.h>

typedef __attribute__((ext_vector_type(16))) _Float16 v16h;
typedef __attribute__((ext_vector_type(8)))  float    v8f;
typedef __attribute__((ext_vector_type(4)))  float    v4f;

#define FEAT 512
#define ATTN 256
#define BATCH 32
#define NSEQ 4096

#define ROW_STRIDE 516   // 512 floats + 4-float pad -> bank step of 4, conflict-free fragments

// gfx1250 hardware tanh (trans op) instead of branchy ocml expansion
__device__ inline float fast_tanh(float x) {
    float y;
    asm("v_tanh_f32 %0, %1" : "=v"(y) : "v"(x));
    return y;
}

// ---- Kernel 1: repack Wf (f32, row-major 512x256) into f16 WMMA B-fragment order.
// wfp[((t*16 + kk)*32 + lane)*16 + j] = Wf[kk*32 + (lane>>4)*8 + (j&7) + (j>>3)*16, t*16 + (lane&15)]
__global__ void pack_wf_kernel(const float* __restrict__ Wf, _Float16* __restrict__ wfp) {
    int idx  = blockIdx.x * 256 + threadIdx.x;     // 131072 total
    int j    = idx & 15;
    int lane = (idx >> 4) & 31;
    int kk   = (idx >> 9) & 15;
    int t    = idx >> 13;
    int k = kk * 32 + ((lane >> 4) << 3) + (j & 7) + ((j >> 3) << 4);
    int n = t * 16 + (lane & 15);
    wfp[idx] = (_Float16)Wf[k * ATTN + n];
}

// ---- Kernel 2: hcomb[b][a] = hidden[b,:]@Wh[:,a] + bh[a] + bf[a]   (32x256, trivial)
__global__ void hcomb_kernel(const float* __restrict__ hidden, const float* __restrict__ Wh,
                             const float* __restrict__ bh, const float* __restrict__ bf,
                             float* __restrict__ hcomb) {
    int idx = blockIdx.x * 256 + threadIdx.x;      // 8192 total
    int b = idx >> 8;
    int a = idx & 255;
    float s = bh[a] + bf[a];
    const float* hp = hidden + b * FEAT;
    for (int k = 0; k < FEAT; ++k) s = fmaf(hp[k], Wh[k * ATTN + a], s);
    hcomb[idx] = s;
}

// ---- Kernel 3: fused GEMM(WMMA f16) + tanh + Ws-dot -> scores (B,N)
// grid (N/32, B), block 256 (8 waves). Block stages a 32x512 feats tile into LDS once via
// async global->LDS copies, then wave w = (rowgroup rg = w>>2, colgroup cg = w&3) computes
// n-tiles {cg, cg+4, cg+8, cg+12} for rows rg*16..rg*16+15.
__global__ __launch_bounds__(256)
void scores_kernel(const float* __restrict__ feats, const _Float16* __restrict__ wfp,
                   const float* __restrict__ hcomb, const float* __restrict__ Ws,
                   const float* __restrict__ bs, float* __restrict__ scores) {
    const int b    = blockIdx.y;
    const int row0 = blockIdx.x * 32;
    const int tid  = threadIdx.x;
    const int w    = tid >> 5;
    const int lane = tid & 31;
    const int m    = lane & 15;
    const int kh   = lane >> 4;          // which K half-group this lane carries
    const int rg   = w >> 2;             // row group 0/1
    const int cg   = w & 3;              // col group 0..3

    __shared__ float tile[32 * ROW_STRIDE];   // padded 32x512 f32 feats tile (~64.5 KB)
    __shared__ float red[8][16];

    // ---- Stage feats tile (contiguous 64 KB in global) into LDS with async b128 copies.
    // 4096 16-byte chunks; thread handles 16 of them (coalesced: wave covers 512B runs).
    {
        const float* gbase = feats + ((size_t)b * NSEQ + row0) * FEAT;
#pragma unroll
        for (int i = 0; i < 16; ++i) {
            int c     = i * 256 + tid;        // chunk id 0..4095
            int row   = c >> 7;               // 128 chunks per row
            int col16 = c & 127;
            uint64_t ga = (uint64_t)(uintptr_t)(gbase + (size_t)row * FEAT + col16 * 4);
            uint32_t la = (uint32_t)(uintptr_t)(tile + row * ROW_STRIDE + col16 * 4);
            asm volatile("global_load_async_to_lds_b128 %0, %1, off"
                         :: "v"(la), "v"(ga) : "memory");
        }
        asm volatile("s_wait_asynccnt 0x0" ::: "memory");
    }
    __syncthreads();

    const v16h* bp = (const v16h*)wfp;
    const float* lrow = tile + (rg * 16 + m) * ROW_STRIDE;

    v8f acc[4];
#pragma unroll
    for (int u = 0; u < 4; ++u) acc[u] = (v8f){};

    for (int kk = 0; kk < 16; ++kk) {
        // A fragment from LDS: lane holds row m of its group, K = {kh*8..+7} and {16+kh*8..+7}
        const float* lp = lrow + kk * 32 + kh * 8;
        v4f f0 = *(const v4f*)(lp);
        v4f f1 = *(const v4f*)(lp + 4);
        v4f f2 = *(const v4f*)(lp + 16);
        v4f f3 = *(const v4f*)(lp + 20);
        v16h a;
        a[0]  = (_Float16)f0[0]; a[1]  = (_Float16)f0[1]; a[2]  = (_Float16)f0[2]; a[3]  = (_Float16)f0[3];
        a[4]  = (_Float16)f1[0]; a[5]  = (_Float16)f1[1]; a[6]  = (_Float16)f1[2]; a[7]  = (_Float16)f1[3];
        a[8]  = (_Float16)f2[0]; a[9]  = (_Float16)f2[1]; a[10] = (_Float16)f2[2]; a[11] = (_Float16)f2[3];
        a[12] = (_Float16)f3[0]; a[13] = (_Float16)f3[1]; a[14] = (_Float16)f3[2]; a[15] = (_Float16)f3[3];

#pragma unroll
        for (int u = 0; u < 4; ++u) {
            v16h bf16 = bp[(((cg + 4 * u) * 16 + kk) * 32) + lane];
            acc[u] = __builtin_amdgcn_wmma_f32_16x16x32_f16(false, a, false, bf16,
                                                            (short)0, acc[u], false, false);
        }
    }

    // ---- Epilogue: tanh(f + hcomb) * Ws, reduce over n.
    float p[8];
#pragma unroll
    for (int r = 0; r < 8; ++r) p[r] = 0.f;
#pragma unroll
    for (int u = 0; u < 4; ++u) {
        const int n   = (cg + 4 * u) * 16 + m;
        const float wsv = Ws[n];
        const float hv  = hcomb[b * ATTN + n];
#pragma unroll
        for (int r = 0; r < 8; ++r)
            p[r] += fast_tanh(acc[u][r] + hv) * wsv;
    }
    // reduce across the 16 lanes sharing a row set (xor 1,2,4,8 stays in half-wave)
#pragma unroll
    for (int r = 0; r < 8; ++r)
#pragma unroll
        for (int off = 1; off < 16; off <<= 1)
            p[r] += __shfl_xor(p[r], off, 32);

    if (m == 0) {
#pragma unroll
        for (int r = 0; r < 8; ++r) red[w][kh * 8 + r] = p[r];  // lane0: rows 0-7, lane16: 8-15
    }
    __syncthreads();
    if (tid < 32) {   // deterministic fixed-order sum over the 4 waves of each row group
        const int rgo = tid >> 4;
        float s = 0.f;
#pragma unroll
        for (int q = 0; q < 4; ++q) s += red[rgo * 4 + q][tid & 15];
        scores[(size_t)b * NSEQ + row0 + tid] = s + bs[0];
    }
}

// ---- Kernel 4: softmax over N per batch -> alpha (written straight into d_out)
__global__ __launch_bounds__(256)
void softmax_kernel(const float* __restrict__ scores, float* __restrict__ alpha) {
    const int b = blockIdx.x;
    const int tid = threadIdx.x;
    __shared__ float sred[256];
    const float* sp = scores + (size_t)b * NSEQ;
    float* ap = alpha + (size_t)b * NSEQ;

    float mx = -3.402823e38f;
    for (int i = tid; i < NSEQ; i += 256) mx = fmaxf(mx, sp[i]);
    sred[tid] = mx; __syncthreads();
    for (int s = 128; s > 0; s >>= 1) { if (tid < s) sred[tid] = fmaxf(sred[tid], sred[tid + s]); __syncthreads(); }
    mx = sred[0]; __syncthreads();

    float sum = 0.f;
    for (int i = tid; i < NSEQ; i += 256) { float e = __expf(sp[i] - mx); ap[i] = e; sum += e; }
    sred[tid] = sum; __syncthreads();
    for (int s = 128; s > 0; s >>= 1) { if (tid < s) sred[tid] += sred[tid + s]; __syncthreads(); }
    float inv = 1.0f / sred[0];
    for (int i = tid; i < NSEQ; i += 256) ap[i] *= inv;
}

// ---- Kernel 5: ctx partials. grid (8, B): block (b,chunk) streams 512 rows of feats.
__global__ __launch_bounds__(256)
void ctx_partial_kernel(const float* __restrict__ feats, const float* __restrict__ alpha,
                        float* __restrict__ part) {
    const int b = blockIdx.y;
    const int chunk = blockIdx.x;   // 0..7
    const int d = threadIdx.x;      // handles d and d+256
    const float* apv = alpha + (size_t)b * NSEQ + chunk * 512;
    const float* fp  = feats + ((size_t)b * NSEQ + chunk * 512) * FEAT;
    float a0 = 0.f, a1 = 0.f;
    for (int n = 0; n < 512; ++n) {
        float al = apv[n];
        a0 = fmaf(al, fp[(size_t)n * FEAT + d], a0);
        a1 = fmaf(al, fp[(size_t)n * FEAT + d + 256], a1);
    }
    float* pp = part + ((size_t)b * 8 + chunk) * FEAT;
    pp[d] = a0; pp[d + 256] = a1;
}

// ---- Kernel 6: reduce 8 chunk-partials -> ctx (d_out front)
__global__ void ctx_reduce_kernel(const float* __restrict__ part, float* __restrict__ ctx) {
    int idx = blockIdx.x * 256 + threadIdx.x;  // 16384
    int b = idx >> 9;
    int d = idx & 511;
    float s = 0.f;
#pragma unroll
    for (int c = 0; c < 8; ++c) s += part[((size_t)b * 8 + c) * FEAT + d];
    ctx[idx] = s;
}

extern "C" void kernel_launch(void* const* d_in, const int* in_sizes, int n_in,
                              void* d_out, int out_size, void* d_ws, size_t ws_size,
                              hipStream_t stream) {
    const float* feats  = (const float*)d_in[0];
    const float* hidden = (const float*)d_in[1];
    const float* Wf     = (const float*)d_in[2];
    const float* bf     = (const float*)d_in[3];
    const float* Wh     = (const float*)d_in[4];
    const float* bh     = (const float*)d_in[5];
    const float* Ws     = (const float*)d_in[6];
    const float* bs     = (const float*)d_in[7];

    float* out   = (float*)d_out;
    float* ctx   = out;                       // 32*512
    float* alpha = out + BATCH * FEAT;        // 32*4096

    char* ws = (char*)d_ws;
    _Float16* wfp   = (_Float16*)ws;                          // 256 KB
    float*    hcomb = (float*)(ws + 262144);                  // 32 KB
    float*    scr   = (float*)(ws + 262144 + 32768);          // 512 KB
    float*    part  = (float*)(ws + 262144 + 32768 + 524288); // 512 KB

    pack_wf_kernel<<<512, 256, 0, stream>>>(Wf, wfp);
    hcomb_kernel<<<32, 256, 0, stream>>>(hidden, Wh, bh, bf, hcomb);
    scores_kernel<<<dim3(NSEQ / 32, BATCH), 256, 0, stream>>>(feats, wfp, hcomb, Ws, bs, scr);
    softmax_kernel<<<BATCH, 256, 0, stream>>>(scr, alpha);
    ctx_partial_kernel<<<dim3(8, BATCH), 256, 0, stream>>>(feats, alpha, part);
    ctx_reduce_kernel<<<64, 256, 0, stream>>>(part, ctx);
}
